// get_model_78271484002340
// MI455X (gfx1250) — matrix operand order repaired and checked by reading, not compiled, not run on hardware
//
#include <hip/hip_runtime.h>
#include <hip/hip_bf16.h>

typedef __attribute__((ext_vector_type(16))) _Float16 v16h;
typedef __attribute__((ext_vector_type(8)))  _Float16 v8h;
typedef __attribute__((ext_vector_type(8)))  float    v8f;

#define BATCH 8
#define NPTS  4096
#define KNN_K 20

// ---------------- split input [B,6,N] -> xyz [B,N,3], nrm [B,N,3] ----------------
__global__ void split_kernel(const float* __restrict__ x, float* __restrict__ xyz,
                             float* __restrict__ nrm) {
  int i = blockIdx.x * blockDim.x + threadIdx.x;
  if (i >= BATCH * NPTS) return;
  int b = i / NPTS, n = i % NPTS;
  const float* xb = x + (size_t)b * 6 * NPTS;
  float* p = xyz + (size_t)i * 3;
  float* q = nrm + (size_t)i * 3;
  p[0] = xb[0 * NPTS + n]; p[1] = xb[1 * NPTS + n]; p[2] = xb[2 * NPTS + n];
  q[0] = xb[3 * NPTS + n]; q[1] = xb[4 * NPTS + n]; q[2] = xb[5 * NPTS + n];
}

// ---------------- furthest point sampling: one block per batch ----------------
__global__ void fps_kernel(const float* __restrict__ pts, int N, int npoint,
                           int* __restrict__ fidx) {
  __shared__ float dist[4096];
  __shared__ float rv[256];
  __shared__ int   ri[256];
  int b = blockIdx.x, tid = threadIdx.x;
  const float* p = pts + (size_t)b * N * 3;
  for (int i = tid; i < N; i += blockDim.x) dist[i] = 1e10f;
  __syncthreads();
  int far = 0;
  for (int s = 0; s < npoint; ++s) {
    if (tid == 0) fidx[b * npoint + s] = far;
    float fx = p[far * 3 + 0], fy = p[far * 3 + 1], fz = p[far * 3 + 2];
    float best = -1.0f; int bi = 0x7fffffff;
    for (int i = tid; i < N; i += blockDim.x) {
      float dx = p[i * 3 + 0] - fx, dy = p[i * 3 + 1] - fy, dz = p[i * 3 + 2] - fz;
      float d = dx * dx + dy * dy + dz * dz;
      float nd = fminf(dist[i], d);
      dist[i] = nd;
      if (nd > best) { best = nd; bi = i; }   // strict > keeps lowest index
    }
    rv[tid] = best; ri[tid] = bi;
    __syncthreads();
    for (int off = 128; off > 0; off >>= 1) {
      if (tid < off) {
        float ov = rv[tid + off]; int oi = ri[tid + off];
        if (ov > rv[tid] || (ov == rv[tid] && oi < ri[tid])) { rv[tid] = ov; ri[tid] = oi; }
      }
      __syncthreads();
    }
    far = ri[0];
    __syncthreads();
  }
}

// ---------------- gather sampled coordinates ----------------
__global__ void gather_xyz_kernel(const float* __restrict__ pts, const int* __restrict__ idx,
                                  int S, int Nsrc, float* __restrict__ out) {
  int i = blockIdx.x * blockDim.x + threadIdx.x;
  if (i >= BATCH * S) return;
  int b = i / S;
  int id = idx[i];
  const float* src = pts + ((size_t)b * Nsrc + id) * 3;
  float* dst = out + (size_t)i * 3;
  dst[0] = src[0]; dst[1] = src[1]; dst[2] = src[2];
}

// ---------------- brute-force kNN (matches top_k(-d) stability via strict <) ----------------
template <int KK>
__global__ void knn_kernel(const float* __restrict__ q, int S,
                           const float* __restrict__ ref, int M,
                           int* __restrict__ out) {
  int i = blockIdx.x * blockDim.x + threadIdx.x;
  if (i >= BATCH * S) return;
  int b = i / S;
  float qx = q[i * 3 + 0], qy = q[i * 3 + 1], qz = q[i * 3 + 2];
  float qq = qx * qx + qy * qy + qz * qz;
  float bd[KK]; int bi[KK];
#pragma unroll
  for (int t = 0; t < KK; ++t) { bd[t] = 1e30f; bi[t] = 0; }
  const float* r = ref + (size_t)b * M * 3;
  for (int j = 0; j < M; ++j) {
    float rx = r[j * 3 + 0], ry = r[j * 3 + 1], rz = r[j * 3 + 2];
    float d = qq - 2.0f * (qx * rx + qy * ry + qz * rz) + (rx * rx + ry * ry + rz * rz);
    if (d < bd[KK - 1]) {
      int p = KK - 1;
#pragma unroll
      for (int t = KK - 1; t > 0; --t) {
        if (d < bd[t - 1]) { bd[t] = bd[t - 1]; bi[t] = bi[t - 1]; p = t - 1; }
      }
      bd[p] = d; bi[p] = j;
    }
  }
#pragma unroll
  for (int t = 0; t < KK; ++t) out[(size_t)i * KK + t] = bi[t];
}

// ---------------- build grouped features in f16, K-padded stride Kp ----------------
__global__ void group_kernel(const float* __restrict__ pts, const float* __restrict__ feats, int C,
                             const int* __restrict__ nidx, const float* __restrict__ new_xyz,
                             int S, int M, int Kp, _Float16* __restrict__ grp) {
  int i = blockIdx.x * blockDim.x + threadIdx.x;   // (b*S+s)*K + kk
  if (i >= BATCH * S * KNN_K) return;
  int bs = i / KNN_K;
  int b = bs / S;
  int id = nidx[i];
  const float* src = pts + ((size_t)b * M + id) * 3;
  const float* nq = new_xyz + (size_t)bs * 3;
  _Float16* dst = grp + (size_t)i * Kp;
  dst[0] = (_Float16)(src[0] - nq[0]);
  dst[1] = (_Float16)(src[1] - nq[1]);
  dst[2] = (_Float16)(src[2] - nq[2]);
  const float* f = feats + ((size_t)b * M + id) * C;
  for (int c = 0; c < C; ++c) dst[3 + c] = (_Float16)f[c];
  for (int c = 3 + C; c < Kp; ++c) dst[c] = (_Float16)0.0f;
}

// ---------------- max-pool over K neighbors ----------------
__global__ void maxpool_kernel(const float* __restrict__ h, int S, int C,
                               float* __restrict__ out) {
  int i = blockIdx.x * blockDim.x + threadIdx.x;   // (b*S+s)*C + c
  if (i >= BATCH * S * C) return;
  int c = i % C; int bs = i / C;
  const float* src = h + (size_t)bs * KNN_K * C + c;
  float m = src[0];
  for (int kk = 1; kk < KNN_K; ++kk) m = fmaxf(m, src[(size_t)kk * C]);
  out[i] = m;
}

// ---------------- FP: 3-NN interp + concat skip -> f16 cat with stride Kp ----------------
__global__ void fp_cat_kernel(const float* __restrict__ xq, int S,
                              const float* __restrict__ xr, int M,
                              const float* __restrict__ fr, int Cr,
                              const int* __restrict__ idx3,
                              const float* __restrict__ skip, int Cs,
                              int Kp, _Float16* __restrict__ cat) {
  int bs = blockIdx.x;                 // b*S + s
  int b = bs / S;
  const int* id = idx3 + (size_t)bs * 3;
  const float* q = xq + (size_t)bs * 3;
  float w0, w1, w2;
  {
    const float* r0 = xr + ((size_t)b * M + id[0]) * 3;
    const float* r1 = xr + ((size_t)b * M + id[1]) * 3;
    const float* r2 = xr + ((size_t)b * M + id[2]) * 3;
    float d0 = (r0[0]-q[0])*(r0[0]-q[0]) + (r0[1]-q[1])*(r0[1]-q[1]) + (r0[2]-q[2])*(r0[2]-q[2]);
    float d1 = (r1[0]-q[0])*(r1[0]-q[0]) + (r1[1]-q[1])*(r1[1]-q[1]) + (r1[2]-q[2])*(r1[2]-q[2]);
    float d2 = (r2[0]-q[0])*(r2[0]-q[0]) + (r2[1]-q[1])*(r2[1]-q[1]) + (r2[2]-q[2])*(r2[2]-q[2]);
    w0 = 1.0f / (d0 + 1e-8f);
    w1 = 1.0f / (d1 + 1e-8f);
    w2 = 1.0f / (d2 + 1e-8f);
    float inv = 1.0f / (w0 + w1 + w2);
    w0 *= inv; w1 *= inv; w2 *= inv;
  }
  const float* f0 = fr + ((size_t)b * M + id[0]) * Cr;
  const float* f1 = fr + ((size_t)b * M + id[1]) * Cr;
  const float* f2 = fr + ((size_t)b * M + id[2]) * Cr;
  _Float16* dst = cat + (size_t)bs * Kp;
  for (int c = threadIdx.x; c < Cr; c += blockDim.x)
    dst[c] = (_Float16)(w0 * f0[c] + w1 * f1[c] + w2 * f2[c]);
  const float* sk = skip + (size_t)bs * Cs;
  for (int c = threadIdx.x; c < Cs; c += blockDim.x)
    dst[Cr + c] = (_Float16)sk[c];
  for (int c = Cr + Cs + threadIdx.x; c < Kp; c += blockDim.x)
    dst[c] = (_Float16)0.0f;
}

// ---------------- skip0 = [cls(16), xyz(3), nrm(3)] per point (f32) ----------------
__global__ void skip0_kernel(const float* __restrict__ cls, const float* __restrict__ xyz,
                             const float* __restrict__ nrm, float* __restrict__ skip) {
  int i = blockIdx.x * blockDim.x + threadIdx.x;   // b*N + n
  if (i >= BATCH * NPTS) return;
  int b = i / NPTS;
  float* d = skip + (size_t)i * 22;
  for (int c = 0; c < 16; ++c) d[c] = cls[b * 16 + c];
  d[16] = xyz[(size_t)i * 3 + 0]; d[17] = xyz[(size_t)i * 3 + 1]; d[18] = xyz[(size_t)i * 3 + 2];
  d[19] = nrm[(size_t)i * 3 + 0]; d[20] = nrm[(size_t)i * 3 + 1]; d[21] = nrm[(size_t)i * 3 + 2];
}

// ---------------- weight prep: W[K,Nc] f32 -> Wt[Ncp][Kp] f16, zero-padded ----------------
__global__ void wprep_kernel(const float* __restrict__ W, int K, int Nc, int Kp, int Ncp,
                             _Float16* __restrict__ out) {
  int i = blockIdx.x * blockDim.x + threadIdx.x;
  if (i >= Ncp * Kp) return;
  int c = i / Kp, k = i % Kp;
  float v = (c < Nc && k < K) ? W[(size_t)k * Nc + c] : 0.0f;
  out[i] = (_Float16)v;
}

// ---------------- f32 -> f16 convert (for g0 feeding the seg GEMM) ----------------
__global__ void cvt_f16_kernel(const float* __restrict__ in, _Float16* __restrict__ out, int n) {
  int i = blockIdx.x * blockDim.x + threadIdx.x;
  if (i >= n) return;
  out[i] = (_Float16)in[i];
}

// ---------------- WMMA GEMM: out[R,Nc] = act(A16[R,Kp] x Wt[Ncp,Kp]^T + bias) ----------------
// 128 threads = 4 wave32; each wave owns one 16x16 output tile along N.
// A16 f16 row-major stride Kp (mult of 32, zero-padded). Wt f16 [Ncp][Kp] (col-major weights).
// Inner loop: 2x b128 A loads + 2x b128 B loads + 1 WMMA. No LDS, no branches.
__global__ void wmma_gemm_kernel(const _Float16* __restrict__ A, int Kp,
                                 const _Float16* __restrict__ Wt,
                                 const float* __restrict__ bias, int Nc,
                                 float* __restrict__ out, int relu) {
  int lane = threadIdx.x & 31;
  int wv   = threadIdx.x >> 5;
  int tileM = blockIdx.x * 16;
  int tileN = (blockIdx.y * 4 + wv) * 16;
  int half = lane >> 4;      // 0: lanes 0-15, 1: lanes 16-31
  int l16  = lane & 15;

  // A frag (16-bit A 16x32): lanes 0-15 row=l16 K 0-7 & 16-23; lanes 16-31 K 8-15 & 24-31
  const _Float16* arow = A  + (size_t)(tileM + l16) * Kp + half * 8;
  // B frag (16-bit B 32x16): N = lane%16; lanes 0-15 K 0-7 & 16-23; lanes 16-31 K 8-15 & 24-31
  const _Float16* bcol = Wt + (size_t)(tileN + l16) * Kp + half * 8;

  v8f acc = {};
  for (int k0 = 0; k0 < Kp; k0 += 32) {
    v8h alo = *(const v8h*)(arow + k0);        // 16B aligned
    v8h ahi = *(const v8h*)(arow + k0 + 16);   // 16B aligned
    v16h a;
#pragma unroll
    for (int e = 0; e < 8; ++e) { a[e] = alo[e]; a[e + 8] = ahi[e]; }
    v8h blo = *(const v8h*)(bcol + k0);        // 16B aligned
    v8h bhi = *(const v8h*)(bcol + k0 + 16);   // 16B aligned
    v16h bf;
#pragma unroll
    for (int e = 0; e < 8; ++e) { bf[e] = blo[e]; bf[e + 8] = bhi[e]; }
    if (k0 + 32 < Kp) __builtin_prefetch(bcol + k0 + 32, 0, 1);
    acc = __builtin_amdgcn_wmma_f32_16x16x32_f16(false, a, false, bf,
                                                 (short)0, acc, false, false);
  }
  // C/D layout: lanes 0-15 M=r N=lane; lanes 16-31 M=8+r N=lane-16
  int col = tileN + l16;
  if (col < Nc) {
    float bv = bias[col];
#pragma unroll
    for (int r = 0; r < 8; ++r) {
      float v = acc[r] + bv;
      if (relu) v = fmaxf(v, 0.0f);
      out[(size_t)(tileM + half * 8 + r) * Nc + col] = v;
    }
  }
}

// ---------------- final transpose g0 [B,N,128] -> [B,128,N] ----------------
__global__ void transpose_g0_kernel(const float* __restrict__ g0, float* __restrict__ out) {
  int i = blockIdx.x * blockDim.x + threadIdx.x;
  if (i >= BATCH * NPTS * 128) return;
  int c = i % 128; int bn = i / 128; int b = bn / NPTS; int n = bn % NPTS;
  out[((size_t)b * 128 + c) * NPTS + n] = g0[i];
}

static inline int roundup(int v, int m) { return (v + m - 1) / m * m; }

extern "C" void kernel_launch(void* const* d_in, const int* in_sizes, int n_in,
                              void* d_out, int out_size, void* d_ws, size_t ws_size,
                              hipStream_t stream) {
  (void)in_sizes; (void)n_in; (void)out_size; (void)ws_size;
  const float* x    = (const float*)d_in[0];
  const float* cls  = (const float*)d_in[1];
  const float* W0 = (const float*)d_in[2];  const float* b0 = (const float*)d_in[3];
  const float* W1 = (const float*)d_in[4];  const float* b1 = (const float*)d_in[5];
  const float* W2 = (const float*)d_in[6];  const float* b2 = (const float*)d_in[7];
  const float* W3 = (const float*)d_in[8];  const float* b3 = (const float*)d_in[9];
  const float* F3 = (const float*)d_in[10]; const float* fb3 = (const float*)d_in[11];
  const float* F2 = (const float*)d_in[12]; const float* fb2 = (const float*)d_in[13];
  const float* F1 = (const float*)d_in[14]; const float* fb1 = (const float*)d_in[15];
  const float* F0 = (const float*)d_in[16]; const float* fb0 = (const float*)d_in[17];
  const float* Wseg = (const float*)d_in[18]; const float* bseg = (const float*)d_in[19];

  char* ws = (char*)d_ws;
  size_t off = 0;
  auto alloc = [&](size_t nbytes) -> void* {
    void* p = ws + off;
    off += (nbytes + 255) & ~(size_t)255;
    return p;
  };
  const int B = BATCH, N = NPTS;

  float* xyz = (float*)alloc((size_t)B * N * 3 * 4);
  float* nrm = (float*)alloc((size_t)B * N * 3 * 4);
  float* x1 = (float*)alloc((size_t)B * 512 * 3 * 4);
  float* f1 = (float*)alloc((size_t)B * 512 * 64 * 4);
  float* x2 = (float*)alloc((size_t)B * 256 * 3 * 4);
  float* f2 = (float*)alloc((size_t)B * 256 * 128 * 4);
  float* x3 = (float*)alloc((size_t)B * 128 * 3 * 4);
  float* f3 = (float*)alloc((size_t)B * 128 * 256 * 4);
  float* x4 = (float*)alloc((size_t)B * 64 * 3 * 4);
  float* f4 = (float*)alloc((size_t)B * 64 * 512 * 4);
  float* gp3 = (float*)alloc((size_t)B * 128 * 512 * 4);
  float* gp2 = (float*)alloc((size_t)B * 256 * 256 * 4);
  float* gp1 = (float*)alloc((size_t)B * 512 * 128 * 4);
  float* gp0 = (float*)alloc((size_t)B * N * 128 * 4);
  _Float16* gp0h = (_Float16*)alloc((size_t)B * N * 128 * 2);
  float* skip0 = (float*)alloc((size_t)B * N * 22 * 4);
  int*   fidx = (int*)alloc((size_t)B * 512 * 4);
  int*   nidx = (int*)alloc((size_t)B * 512 * KNN_K * 4);
  int*   idx3 = (int*)alloc((size_t)B * N * 3 * 4);
  _Float16* catA = (_Float16*)alloc((size_t)B * N * 160 * 2);            // max f16 A buffer
  float* hbuf = (float*)alloc((size_t)B * 512 * KNN_K * 64 * 4);         // max GEMM-out f32

  // K-pad / N-pad sizes per layer
  const int Kp0 = roundup(6, 32),   Np0 = roundup(64, 64);
  const int Kp1 = roundup(67, 32),  Np1 = roundup(128, 64);
  const int Kp2 = roundup(131, 32), Np2 = roundup(256, 64);
  const int Kp3 = roundup(259, 32), Np3 = roundup(512, 64);
  const int KpF3 = roundup(768, 32), NpF3 = roundup(512, 64);
  const int KpF2 = roundup(640, 32), NpF2 = roundup(256, 64);
  const int KpF1 = roundup(320, 32), NpF1 = roundup(128, 64);
  const int KpF0 = roundup(150, 32), NpF0 = roundup(128, 64);
  const int KpS  = 128,              NpS  = roundup(50, 64);

  _Float16* tW0 = (_Float16*)alloc((size_t)Np0 * Kp0 * 2);
  _Float16* tW1 = (_Float16*)alloc((size_t)Np1 * Kp1 * 2);
  _Float16* tW2 = (_Float16*)alloc((size_t)Np2 * Kp2 * 2);
  _Float16* tW3 = (_Float16*)alloc((size_t)Np3 * Kp3 * 2);
  _Float16* tF3 = (_Float16*)alloc((size_t)NpF3 * KpF3 * 2);
  _Float16* tF2 = (_Float16*)alloc((size_t)NpF2 * KpF2 * 2);
  _Float16* tF1 = (_Float16*)alloc((size_t)NpF1 * KpF1 * 2);
  _Float16* tF0 = (_Float16*)alloc((size_t)NpF0 * KpF0 * 2);
  _Float16* tWs = (_Float16*)alloc((size_t)NpS * KpS * 2);

  float* out_res = (float*)d_out;                 // [B,N,50]
  float* out_g0t = out_res + (size_t)B * N * 50;  // [B,128,N]

  dim3 blk256(256);
#define WPREP(Wp, K, Nc, Kp, Ncp, dst) \
  wprep_kernel<<<dim3(((Ncp) * (Kp) + 255) / 256), blk256, 0, stream>>>(Wp, K, Nc, Kp, Ncp, dst)
  WPREP(W0, 6, 64, Kp0, Np0, tW0);
  WPREP(W1, 67, 128, Kp1, Np1, tW1);
  WPREP(W2, 131, 256, Kp2, Np2, tW2);
  WPREP(W3, 259, 512, Kp3, Np3, tW3);
  WPREP(F3, 768, 512, KpF3, NpF3, tF3);
  WPREP(F2, 640, 256, KpF2, NpF2, tF2);
  WPREP(F1, 320, 128, KpF1, NpF1, tF1);
  WPREP(F0, 150, 128, KpF0, NpF0, tF0);
  WPREP(Wseg, 128, 50, KpS, NpS, tWs);
#undef WPREP

  split_kernel<<<dim3((B * N + 255) / 256), blk256, 0, stream>>>(x, xyz, nrm);

  // ---- SA0: 4096 -> 512, MLP 6 -> 64
  fps_kernel<<<dim3(B), blk256, 0, stream>>>(xyz, N, 512, fidx);
  gather_xyz_kernel<<<dim3((B * 512 + 255) / 256), blk256, 0, stream>>>(xyz, fidx, 512, N, x1);
  knn_kernel<KNN_K><<<dim3((B * 512 + 255) / 256), blk256, 0, stream>>>(x1, 512, xyz, N, nidx);
  group_kernel<<<dim3((B * 512 * KNN_K + 255) / 256), blk256, 0, stream>>>(xyz, nrm, 3, nidx, x1, 512, N, Kp0, catA);
  wmma_gemm_kernel<<<dim3(B * 512 * KNN_K / 16, Np0 / 64), dim3(128), 0, stream>>>(catA, Kp0, tW0, b0, 64, hbuf, 1);
  maxpool_kernel<<<dim3((B * 512 * 64 + 255) / 256), blk256, 0, stream>>>(hbuf, 512, 64, f1);

  // ---- SA1: 512 -> 256, MLP 67 -> 128
  fps_kernel<<<dim3(B), blk256, 0, stream>>>(x1, 512, 256, fidx);
  gather_xyz_kernel<<<dim3((B * 256 + 255) / 256), blk256, 0, stream>>>(x1, fidx, 256, 512, x2);
  knn_kernel<KNN_K><<<dim3((B * 256 + 255) / 256), blk256, 0, stream>>>(x2, 256, x1, 512, nidx);
  group_kernel<<<dim3((B * 256 * KNN_K + 255) / 256), blk256, 0, stream>>>(x1, f1, 64, nidx, x2, 256, 512, Kp1, catA);
  wmma_gemm_kernel<<<dim3(B * 256 * KNN_K / 16, Np1 / 64), dim3(128), 0, stream>>>(catA, Kp1, tW1, b1, 128, hbuf, 1);
  maxpool_kernel<<<dim3((B * 256 * 128 + 255) / 256), blk256, 0, stream>>>(hbuf, 256, 128, f2);

  // ---- SA2: 256 -> 128, MLP 131 -> 256
  fps_kernel<<<dim3(B), blk256, 0, stream>>>(x2, 256, 128, fidx);
  gather_xyz_kernel<<<dim3((B * 128 + 255) / 256), blk256, 0, stream>>>(x2, fidx, 128, 256, x3);
  knn_kernel<KNN_K><<<dim3((B * 128 + 255) / 256), blk256, 0, stream>>>(x3, 128, x2, 256, nidx);
  group_kernel<<<dim3((B * 128 * KNN_K + 255) / 256), blk256, 0, stream>>>(x2, f2, 128, nidx, x3, 128, 256, Kp2, catA);
  wmma_gemm_kernel<<<dim3(B * 128 * KNN_K / 16, Np2 / 64), dim3(128), 0, stream>>>(catA, Kp2, tW2, b2, 256, hbuf, 1);
  maxpool_kernel<<<dim3((B * 128 * 256 + 255) / 256), blk256, 0, stream>>>(hbuf, 128, 256, f3);

  // ---- SA3: 128 -> 64, MLP 259 -> 512
  fps_kernel<<<dim3(B), blk256, 0, stream>>>(x3, 128, 64, fidx);
  gather_xyz_kernel<<<dim3((B * 64 + 255) / 256), blk256, 0, stream>>>(x3, fidx, 64, 128, x4);
  knn_kernel<KNN_K><<<dim3((B * 64 + 255) / 256), blk256, 0, stream>>>(x4, 64, x3, 128, nidx);
  group_kernel<<<dim3((B * 64 * KNN_K + 255) / 256), blk256, 0, stream>>>(x3, f3, 256, nidx, x4, 64, 128, Kp3, catA);
  wmma_gemm_kernel<<<dim3(B * 64 * KNN_K / 16, Np3 / 64), dim3(128), 0, stream>>>(catA, Kp3, tW3, b3, 512, hbuf, 1);
  maxpool_kernel<<<dim3((B * 64 * 512 + 255) / 256), blk256, 0, stream>>>(hbuf, 64, 512, f4);

  // ---- FP3: interp f4 @ x3, skip f3, MLP 768 -> 512
  knn_kernel<3><<<dim3((B * 128 + 255) / 256), blk256, 0, stream>>>(x3, 128, x4, 64, idx3);
  fp_cat_kernel<<<dim3(B * 128), dim3(128), 0, stream>>>(x3, 128, x4, 64, f4, 512, idx3, f3, 256, KpF3, catA);
  wmma_gemm_kernel<<<dim3(B * 128 / 16, NpF3 / 64), dim3(128), 0, stream>>>(catA, KpF3, tF3, fb3, 512, gp3, 1);

  // ---- FP2: interp gp3 @ x2, skip f2, MLP 640 -> 256
  knn_kernel<3><<<dim3((B * 256 + 255) / 256), blk256, 0, stream>>>(x2, 256, x3, 128, idx3);
  fp_cat_kernel<<<dim3(B * 256), dim3(128), 0, stream>>>(x2, 256, x3, 128, gp3, 512, idx3, f2, 128, KpF2, catA);
  wmma_gemm_kernel<<<dim3(B * 256 / 16, NpF2 / 64), dim3(128), 0, stream>>>(catA, KpF2, tF2, fb2, 256, gp2, 1);

  // ---- FP1: interp gp2 @ x1, skip f1, MLP 320 -> 128
  knn_kernel<3><<<dim3((B * 512 + 255) / 256), blk256, 0, stream>>>(x1, 512, x2, 256, idx3);
  fp_cat_kernel<<<dim3(B * 512), dim3(128), 0, stream>>>(x1, 512, x2, 256, gp2, 256, idx3, f1, 64, KpF1, catA);
  wmma_gemm_kernel<<<dim3(B * 512 / 16, NpF1 / 64), dim3(128), 0, stream>>>(catA, KpF1, tF1, fb1, 128, gp1, 1);

  // ---- FP0: interp gp1 @ xyz, skip [cls,xyz,nrm], MLP 150 -> 128
  knn_kernel<3><<<dim3((B * N + 255) / 256), blk256, 0, stream>>>(xyz, N, x1, 512, idx3);
  skip0_kernel<<<dim3((B * N + 255) / 256), blk256, 0, stream>>>(cls, xyz, nrm, skip0);
  fp_cat_kernel<<<dim3(B * N), dim3(128), 0, stream>>>(xyz, N, x1, 512, gp1, 128, idx3, skip0, 22, KpF0, catA);
  wmma_gemm_kernel<<<dim3(B * N / 16, NpF0 / 64), dim3(128), 0, stream>>>(catA, KpF0, tF0, fb0, 128, gp0, 1);

  // ---- segmentation head: [B*N,128] x [128,50] + bseg (no relu) -> d_out
  cvt_f16_kernel<<<dim3((B * N * 128 + 255) / 256), blk256, 0, stream>>>(gp0, gp0h, B * N * 128);
  wmma_gemm_kernel<<<dim3(B * N / 16, NpS / 64), dim3(128), 0, stream>>>(gp0h, KpS, tWs, bseg, 50, out_res, 0);

  // ---- second output: transpose(g0) [B,128,N]
  transpose_g0_kernel<<<dim3((B * N * 128 + 255) / 256), blk256, 0, stream>>>(gp0, out_g0t);
}
